// EqualizedModulatedConv2d_26414048870683
// MI455X (gfx1250) — matrix operand order repaired and checked by reading, not compiled, not run on hardware
//
#include <hip/hip_runtime.h>
#include <hip/hip_bf16.h>

typedef __attribute__((ext_vector_type(16))) __bf16 v16bf;
typedef __attribute__((ext_vector_type(8)))  float  v8f;
typedef __attribute__((ext_vector_type(4)))  int    v4i;

#define N_    16
#define IC_   512
#define OC_   512
#define SDIM_ 512
#define H_    32
#define W_    32
#define HW_   (H_*W_)

#define FC_SCALER_     0.04419417382415922f   /* 1/sqrt(512)  */
#define WEIGHT_SCALER_ 0.014731391274719739f  /* 1/sqrt(4608) */

/* ---------------- CDNA5 async global->LDS support (probe-guarded) ----------------- */
#if defined(__has_builtin)
#  if __has_builtin(__builtin_amdgcn_global_load_async_to_lds_b128)
#    define HAVE_ASYNC_LDS 1
#  endif
#  if __has_builtin(__builtin_amdgcn_s_wait_asynccnt)
#    define HAVE_WAIT_ASYNC 1
#  endif
#endif

typedef __attribute__((address_space(1))) v4i gv4i;   /* global int4 */
typedef __attribute__((address_space(3))) v4i lv4i;   /* LDS int4    */

__device__ __forceinline__ void async_copy16(const __bf16* g, __bf16* l) {
#if defined(HAVE_ASYNC_LDS)
  /* global_load_async_to_lds_b128: per-lane global src -> per-lane LDS dst (ASYNCcnt) */
  __builtin_amdgcn_global_load_async_to_lds_b128(
      (gv4i*)(g), (lv4i*)(l), 0, 0);
#else
  *(uint4*)l = *(const uint4*)g;
#endif
}

__device__ __forceinline__ void wait_async_lds() {
#if defined(HAVE_ASYNC_LDS)
#  if defined(HAVE_WAIT_ASYNC)
  __builtin_amdgcn_s_wait_asynccnt(0);
#  else
  asm volatile("s_wait_asynccnt 0x0" ::: "memory");
#  endif
#endif
}

/* ---------------- Stage 1: mod[n,ic] = style @ fc_w^T * FC + bias + 1 ------------- */
__global__ void fc_mod_kernel(const float* __restrict__ style,
                              const float* __restrict__ fcw,
                              const float* __restrict__ fcb,
                              float* __restrict__ mod) {
  int idx = blockIdx.x * blockDim.x + threadIdx.x;      // n*IC + ic
  int n = idx / IC_, ic = idx % IC_;
  const float* s = style + n * SDIM_;
  const float* w = fcw + (size_t)ic * SDIM_;
  float acc = 0.f;
  #pragma unroll 4
  for (int k = 0; k < SDIM_; ++k) acc += s[k] * w[k];
  mod[idx] = acc * FC_SCALER_ + fcb[ic] + 1.0f;
}

/* -------- Stage 2: wsq[oc,ic] = sum_k w^2 ; Wb[k][oc][ic] = bf16(weight) ---------- */
__global__ void prep_weight_kernel(const float* __restrict__ weight,
                                   float* __restrict__ wsq,
                                   __bf16* __restrict__ wb) {
  int idx = blockIdx.x * blockDim.x + threadIdx.x;      // oc*IC + ic
  const float* w = weight + (size_t)idx * 9;
  float s = 0.f;
  #pragma unroll
  for (int k = 0; k < 9; ++k) {
    float v = w[k];
    s += v * v;
    wb[(size_t)k * (OC_ * IC_) + idx] = (__bf16)v;
  }
  wsq[idx] = s;
}

/* -------- Stage 3: scale[n,oc] = WS * rsqrt(WS^2 * sum_ic mod^2*wsq + 1e-8) ------- */
__global__ void demod_kernel(const float* __restrict__ mod,
                             const float* __restrict__ wsq,
                             float* __restrict__ scale) {
  int idx = blockIdx.x * blockDim.x + threadIdx.x;      // n*OC + oc
  int n = idx / OC_, oc = idx % OC_;
  const float* m = mod + n * IC_;
  const float* q = wsq + (size_t)oc * IC_;
  float acc = 0.f;
  #pragma unroll 4
  for (int ic = 0; ic < IC_; ++ic) { float mm = m[ic]; acc += mm * mm * q[ic]; }
  const float ws2 = WEIGHT_SCALER_ * WEIGHT_SCALER_;
  scale[idx] = rsqrtf(ws2 * acc + 1e-8f) * WEIGHT_SCALER_;
}

/* -------- Stage 4: xs[n][h][w][ic] = bf16(x[n][ic][h][w] * mod[n,ic]) (NHWC) ------ */
__global__ void scale_x_kernel(const float* __restrict__ x,
                               const float* __restrict__ mod,
                               __bf16* __restrict__ xs) {
  int idx = blockIdx.x * blockDim.x + threadIdx.x;      // ((n*HW + p)*IC + ic)
  int ic = idx % IC_;
  int t  = idx / IC_;
  int p  = t % HW_;
  int n  = t / HW_;
  float v = x[((size_t)(n * IC_ + ic)) * HW_ + p] * mod[n * IC_ + ic];
  xs[idx] = (__bf16)v;
}

/* ---------------- Stage 5: implicit-GEMM conv, bf16 WMMA, double-buffered --------- */
/* C[oc, pix] += Wb_k[oc, ic] * xs[pix(shifted), ic]  summed over 9 taps, K=IC each   */
#define BM 128            /* oc tile per workgroup     */
#define BN 128            /* pixel tile per workgroup  */
#define BK 64             /* K halves staged per chunk (2 WMMA K-steps) */
#define LDSW 72           /* padded LDS row stride in halves (144B, 16B aligned) */
#define NCHUNK (9 * (IC_ / BK))   /* 72 pipeline stages */

__launch_bounds__(256, 1)
__global__ void conv_wmma_kernel(const __bf16* __restrict__ wb,    /* [9][OC][IC]  */
                                 const __bf16* __restrict__ xs,    /* [N][HW][IC]  */
                                 const float*  __restrict__ scale, /* [N][OC]      */
                                 float* __restrict__ out) {        /* [N][OC][HW]  */
  __shared__ __align__(16) __bf16 lA[2 * BM * LDSW];  /* weight tiles  */
  __shared__ __align__(16) __bf16 lB[2 * BN * LDSW];  /* activ tiles   */

  const int tid  = threadIdx.x;
  const int lane = tid & 31;
  const int wv   = tid >> 5;
  const int wm   = wv & 3;        /* 4 waves along M(oc): 32 rows each  */
  const int wn   = wv >> 2;       /* 2 waves along N(pix): 64 cols each */
  const int hl   = lane >> 4;     /* half-wave select */
  const int l16  = lane & 15;

  const int ocBase = blockIdx.x * BM;
  const int pBase  = blockIdx.y * BN;
  const int n      = blockIdx.z;

  /* ---- loop-invariant fill addressing: 8 x 16B segments per 128x128B tile row ---- */
  const int part8 = (tid & 7) * 8;       /* half offset within a row */
  const int r0    = tid >> 3;            /* first of 4 rows handled by this thread */
  size_t aGOff[4];                       /* global half-offset within a Wb tap plane */
  int    ldsOff[4];                      /* LDS half-offset (same for A and B tiles) */
  int    pY[4], pX[4];                   /* pixel coords for the B rows */
  #pragma unroll
  for (int ld = 0; ld < 4; ++ld) {
    const int row = r0 + 32 * ld;
    aGOff[ld]  = (size_t)(ocBase + row) * IC_ + part8;
    ldsOff[ld] = row * LDSW + part8;
    const int p = pBase + row;
    pY[ld] = p >> 5;
    pX[ld] = p & 31;
  }

  v8f c[2][4];
  #pragma unroll
  for (int i = 0; i < 2; ++i)
    #pragma unroll
    for (int j = 0; j < 4; ++j)
      c[i][j] = (v8f){0.f,0.f,0.f,0.f,0.f,0.f,0.f,0.f};

  /* ---- stage one chunk (16KB A + 16KB B) into LDS buffer `sel` ------------------- */
  auto stage = [&](int chunk, int sel) {
    const int k9 = chunk >> 3;
    const int kc = (chunk & 7) * BK;
    const int dy = k9 / 3 - 1;
    const int dx = k9 % 3 - 1;
    const __bf16* wbk = wb + (size_t)k9 * (OC_ * IC_) + kc;
    const __bf16* xsn = xs + (size_t)n * HW_ * IC_ + kc + part8;
    __bf16* LA = lA + sel * (BM * LDSW);
    __bf16* LB = lB + sel * (BN * LDSW);
    #pragma unroll
    for (int ld = 0; ld < 4; ++ld)
      async_copy16(wbk + aGOff[ld], &LA[ldsOff[ld]]);
    #pragma unroll
    for (int ld = 0; ld < 4; ++ld) {
      const int y = pY[ld] + dy;
      const int x = pX[ld] + dx;
      __bf16* dst = &LB[ldsOff[ld]];
      if ((unsigned)y < 32u && (unsigned)x < 32u) {
        async_copy16(xsn + (size_t)(y * 32 + x) * IC_, dst);
      } else {
        uint4 z = {0u, 0u, 0u, 0u};
        *(uint4*)dst = z;                 /* zero padding via ds_store */
      }
    }
  };

  /* ---- 16 x v_wmma_f32_16x16x32_bf16 on LDS buffer `sel` ------------------------- */
  auto compute = [&](int sel) {
    const __bf16* LA = lA + sel * (BM * LDSW);
    const __bf16* LB = lB + sel * (BN * LDSW);
    #pragma unroll
    for (int ks = 0; ks < 2; ++ks) {
      v16bf a[2], b[4];
      #pragma unroll
      for (int i = 0; i < 2; ++i) {       /* A: lane M=l16; K halves {hl*8, 16+hl*8} */
        const __bf16* r = &LA[(wm * 32 + i * 16 + l16) * LDSW + ks * 32];
        union { v16bf v; uint4 q[2]; } u;
        u.q[0] = *(const uint4*)(r + hl * 8);
        u.q[1] = *(const uint4*)(r + 16 + hl * 8);
        a[i] = u.v;
      }
      #pragma unroll
      for (int j = 0; j < 4; ++j) {       /* B: lane N=l16; K halves {hl*8, 16+hl*8} */
        const __bf16* r = &LB[(wn * 64 + j * 16 + l16) * LDSW + ks * 32];
        union { v16bf v; uint4 q[2]; } u;
        u.q[0] = *(const uint4*)(r + hl * 8);
        u.q[1] = *(const uint4*)(r + 16 + hl * 8);
        b[j] = u.v;
      }
      #pragma unroll
      for (int i = 0; i < 2; ++i)
        #pragma unroll
        for (int j = 0; j < 4; ++j)
          c[i][j] = __builtin_amdgcn_wmma_f32_16x16x32_bf16(
              false, a[i], false, b[j], (short)0, c[i][j], false, false);
    }
  };

  /* ---- software pipeline: fill next buffer while computing current --------------- */
  stage(0, 0);
  wait_async_lds();
  __syncthreads();
  for (int ch = 0; ch < NCHUNK; ++ch) {
    if (ch + 1 < NCHUNK) stage(ch + 1, (ch + 1) & 1);
    compute(ch & 1);
    wait_async_lds();
    __syncthreads();
  }

  /* ---- epilogue: out[n][oc][pix] = c * scale[n][oc]; lanes -> contiguous pix ----- */
  #pragma unroll
  for (int i = 0; i < 2; ++i) {
    #pragma unroll
    for (int r = 0; r < 8; ++r) {
      int oc = ocBase + wm * 32 + i * 16 + r + hl * 8;
      float sc = scale[n * OC_ + oc];
      #pragma unroll
      for (int j = 0; j < 4; ++j) {
        int p = pBase + wn * 64 + j * 16 + l16;
        out[((size_t)(n * OC_ + oc)) * HW_ + p] = c[i][j][r] * sc;
      }
    }
  }
}

/* --------------------------------- launcher --------------------------------------- */
extern "C" void kernel_launch(void* const* d_in, const int* in_sizes, int n_in,
                              void* d_out, int out_size, void* d_ws, size_t ws_size,
                              hipStream_t stream) {
  (void)in_sizes; (void)n_in; (void)out_size; (void)ws_size;
  const float* x    = (const float*)d_in[0];   /* [16,512,32,32] */
  const float* sty  = (const float*)d_in[1];   /* [16,512]       */
  const float* wgt  = (const float*)d_in[2];   /* [512,512,3,3]  */
  const float* fcw  = (const float*)d_in[3];   /* [512,512]      */
  const float* fcb  = (const float*)d_in[4];   /* [512]          */
  float* out = (float*)d_out;                  /* [16,512,32,32] */

  char* ws = (char*)d_ws;
  float*  mod   = (float*) (ws + 0);                         /* 32 KB   */
  float*  scl   = (float*) (ws + 32768);                     /* 32 KB   */
  float*  wsq   = (float*) (ws + 65536);                     /* 1 MB    */
  __bf16* wb    = (__bf16*)(ws + 65536 + 1048576);           /* 4.5 MB  */
  __bf16* xsb   = (__bf16*)(ws + 65536 + 1048576 + 4718592); /* 16 MB   */

  fc_mod_kernel     <<<(N_*IC_)/256,       256, 0, stream>>>(sty, fcw, fcb, mod);
  prep_weight_kernel<<<(OC_*IC_)/256,      256, 0, stream>>>(wgt, wsq, wb);
  demod_kernel      <<<(N_*OC_)/256,       256, 0, stream>>>(mod, wsq, scl);
  scale_x_kernel    <<<(N_*HW_*IC_)/256,   256, 0, stream>>>(x, mod, xsb);

  dim3 grid(OC_/BM, HW_/BN, N_);           /* 4 x 8 x 16 = 512 workgroups */
  conv_wmma_kernel  <<<grid, 256, 0, stream>>>(wb, xsb, scl, out);
}
